// DriverModel_88484916232431
// MI455X (gfx1250) — matrix operand order repaired and checked by reading, not compiled, not run on hardware
//
#include <hip/hip_runtime.h>
#include <hip/hip_bf16.h>

// ============================================================================
// MobileNetV3-Small forward + QP head for MI455X (gfx1250).
//  - activations: NHWC f16, channel dim zero-padded to multiple of 32 (WMMA K)
//  - 1x1 convs: implicit GEMM, v_wmma_f32_16x16x32_f16; 16x CinPad weight tile
//    staged in LDS via TDM (tensor_load_to_lds), A-frags via ds_load_b128,
//    B-frags as unconditional clamped global b128 loads (tail columns are
//    discarded at the guarded store), 2 pixel-tiles per wave (A reuse).
//  - depthwise / stem: direct VALU conv (NHWC, lane-per-channel => coalesced)
//  - train-mode BN: per-channel stats kernel + fused normalize->act->f16-pack
//  - QP: single-thread PDIPM (n=m=8), 20 fixed iterations
// ============================================================================

typedef __attribute__((ext_vector_type(16))) _Float16 v16h;
typedef __attribute__((ext_vector_type(8)))  _Float16 v8h;
typedef __attribute__((ext_vector_type(8)))  float    v8f;
typedef __attribute__((ext_vector_type(4)))  unsigned int v4u;
typedef __attribute__((ext_vector_type(8)))  int      v8i;
typedef __attribute__((ext_vector_type(4)))  int      v4i;

#define HAS_TDM __has_builtin(__builtin_amdgcn_tensor_load_to_lds)

__device__ __forceinline__ float apply_act(float v, int act) {
  if (act == 1) return fmaxf(v, 0.f);                                        // relu
  if (act == 2) return v * fminf(fmaxf(v + 3.f, 0.f), 6.f) * (1.f / 6.f);    // hswish
  if (act == 3) return fminf(fmaxf(v + 3.f, 0.f), 6.f) * (1.f / 6.f);        // hsigmoid
  return v;
}

// ---------------------------------------------------------------------------
// rescale: Conv2d(3,3,(1,2)) valid.  img NCHW [3][768][1537] -> y NHWC [768*1536][3]
// ---------------------------------------------------------------------------
__global__ void k_rescale(const float* __restrict__ img, const float* __restrict__ w,
                          const float* __restrict__ b, float* __restrict__ y, int n) {
  int i = blockIdx.x * blockDim.x + threadIdx.x;
  if (i >= n) return;
  int c  = i % 3;
  int px = i / 3;
  int x  = px % 1536;
  int yy = px / 1536;
  float acc = b[c];
#pragma unroll
  for (int ci = 0; ci < 3; ++ci) {
    const float* row = img + ((size_t)ci * 768 + yy) * 1537 + x;
    acc += row[0] * w[c * 6 + ci * 2 + 0] + row[1] * w[c * 6 + ci * 2 + 1];
  }
  y[i] = acc;
}

// ---------------------------------------------------------------------------
// stem: 3x3 s2 pad1, 3->16. NHWC in [768*1536][3] -> NHWC f32 out [384*768][16]
// ---------------------------------------------------------------------------
__global__ void k_stem(const float* __restrict__ x, const float* __restrict__ w,
                       float* __restrict__ y, int n) {
  int i = blockIdx.x * blockDim.x + threadIdx.x;
  if (i >= n) return;
  int co = i % 16;
  int px = i / 16;
  int wo = px % 768;
  int ho = px / 768;
  float acc = 0.f;
#pragma unroll
  for (int ky = 0; ky < 3; ++ky) {
    int yy = ho * 2 - 1 + ky;
    if ((unsigned)yy >= 768u) continue;
#pragma unroll
    for (int kx = 0; kx < 3; ++kx) {
      int xx = wo * 2 - 1 + kx;
      if ((unsigned)xx >= 1536u) continue;
      const float* pix = x + ((size_t)yy * 1536 + xx) * 3;
#pragma unroll
      for (int ci = 0; ci < 3; ++ci)
        acc += pix[ci] * w[((co * 3 + ci) * 3 + ky) * 3 + kx];
    }
  }
  y[i] = acc;
}

// ---------------------------------------------------------------------------
// weights f32 [Cout][Cin] -> f16 padded [Cout16][CinPad] (zero fill)
// ---------------------------------------------------------------------------
__global__ void k_w2h(const float* __restrict__ W, _Float16* __restrict__ Wh,
                      int Cout, int Cin, int CinPad, int n) {
  int i = blockIdx.x * blockDim.x + threadIdx.x;
  if (i >= n) return;
  int k = i % CinPad;
  int r = i / CinPad;
  Wh[i] = (r < Cout && k < Cin) ? (_Float16)W[(size_t)r * Cin + k] : (_Float16)0.f;
}

// ---------------------------------------------------------------------------
// Pointwise conv, implicit GEMM:  Y[p][co] = sum_k Wh[co][k] * X[p][k]
//   X : f16 NHWC, row stride CinPad (mult of 32, zero padded)
//   Wh: f16 [Cout16][CinPad]
//   Y : f32 NHWC, row stride Cout
// One wave -> 16 (co) x 32 (pixel) = two 16x16 D tiles sharing the A fragment.
// Block = 8 waves -> 16 co x 256 pixels. Weight tile (16 x CinPad) in LDS (TDM).
// ---------------------------------------------------------------------------
__global__ __launch_bounds__(256) void k_gemm_wmma(
    const _Float16* __restrict__ X, const _Float16* __restrict__ Wh,
    float* __restrict__ Y, int CinPad, int Cout, int HW) {
  __shared__ _Float16 wtile[16 * 576];

  int wave   = threadIdx.x >> 5;
  int lane   = threadIdx.x & 31;
  int half16 = lane >> 4;
  int l15    = lane & 15;
  int p0  = blockIdx.x * 256 + wave * 32;
  int co0 = blockIdx.y * 16;

  // ---- stage 16 x CinPad weight tile into LDS (TDM) --------------------------
#if HAS_TDM
  if (threadIdx.x == 0) {
    unsigned long long ga = (unsigned long long)(const void*)(Wh + (size_t)co0 * CinPad);
    unsigned int lds_off =
        (unsigned int)(unsigned long long)(__attribute__((address_space(3))) _Float16*)wtile;
    v4u g0;
    g0[0] = 1u;                                   // count=1, user descriptor
    g0[1] = lds_off;                              // lds_addr (bytes)
    g0[2] = (unsigned int)(ga & 0xffffffffull);   // global_addr[31:0]
    g0[3] = (unsigned int)((ga >> 32) & 0x01ffffffull) | (2u << 30);  // addr[56:32] | type=2
    v8i g1;
    g1[0] = 1 << 16;                              // workgroup_mask=0, data_size=1 (2B)
    g1[1] = (CinPad & 0xffff) << 16;              // tensor_dim0[15:0]
    g1[2] = (16 << 16);                           // tensor_dim0 hi=0 | tensor_dim1 lo=16
    g1[3] = (CinPad << 16);                       // tensor_dim1 hi=0 | tile_dim0=CinPad
    g1[4] = 16;                                   // tile_dim1=16, tile_dim2=0
    g1[5] = CinPad;                               // tensor_dim0_stride lo32
    g1[6] = 0;
    g1[7] = 0;
    v4i zz = {};
    v8i zz8 = {};
    __builtin_amdgcn_tensor_load_to_lds(g0, g1, zz, zz, zz8, 0);
    __builtin_amdgcn_s_wait_tensorcnt(0);
  }
#else
  for (int t = threadIdx.x; t < 16 * CinPad; t += 256)
    wtile[t] = Wh[(size_t)co0 * CinPad + t];
#endif
  __syncthreads();

  // two pixel columns per lane; clamp to stay in-bounds (tail cols discarded)
  int pcol0 = p0 + l15;
  int pcol1 = p0 + 16 + l15;
  int pc0c = pcol0 < HW ? pcol0 : HW - 1;
  int pc1c = pcol1 < HW ? pcol1 : HW - 1;
  const _Float16* xrow0 = X + (size_t)pc0c * CinPad;
  const _Float16* xrow1 = X + (size_t)pc1c * CinPad;

  union H16 { v16h v; v8h h[2]; };
  v8f acc0 = {}, acc1 = {};

  for (int k0 = 0; k0 < CinPad; k0 += 32) {
    // A (16x32 f16): lanes 0-15 hold K {0..7,16..23}, lanes 16-31 K {8..15,24..31}
    const v8h* ap = (const v8h*)(wtile + l15 * CinPad + k0 + half16 * 8);
    H16 a;
    a.h[0] = ap[0];          // K k0+half*8 .. +7
    a.h[1] = ap[2];          // K k0+16+half*8 .. +7
    // B (32x16 f16): lanes 0-15 hold K {0..7,16..23}, lanes 16-31 K {8..15,24..31}
    const v8h* bp0 = (const v8h*)(xrow0 + k0 + half16 * 8);
    const v8h* bp1 = (const v8h*)(xrow1 + k0 + half16 * 8);
    H16 b0, b1;
    b0.h[0] = bp0[0];
    b0.h[1] = bp0[2];
    b1.h[0] = bp1[0];
    b1.h[1] = bp1[2];
    __builtin_prefetch(xrow0 + k0 + 64, 0, 3);    // K-step after next (k0+64) of this pixel's row
    acc0 = __builtin_amdgcn_wmma_f32_16x16x32_f16(false, a.v, false, b0.v, (short)0,
                                                  acc0, false, false);
    acc1 = __builtin_amdgcn_wmma_f32_16x16x32_f16(false, a.v, false, b1.v, (short)0,
                                                  acc1, false, false);
  }
  // D (f32 16x16): lane col = pcol, rows co0+half*8 .. +7 -> contiguous in NHWC
  int cobase = co0 + half16 * 8;
  if (cobase < Cout) {   // Cout mult of 8 -> uniform per 8-row group
    if (pcol0 < HW) {
      float* yp = Y + (size_t)pcol0 * Cout + cobase;
      *(float4*)yp       = make_float4(acc0[0], acc0[1], acc0[2], acc0[3]);
      *(float4*)(yp + 4) = make_float4(acc0[4], acc0[5], acc0[6], acc0[7]);
    }
    if (pcol1 < HW) {
      float* yp = Y + (size_t)pcol1 * Cout + cobase;
      *(float4*)yp       = make_float4(acc1[0], acc1[1], acc1[2], acc1[3]);
      *(float4*)(yp + 4) = make_float4(acc1[4], acc1[5], acc1[6], acc1[7]);
    }
  }
}

// ---------------------------------------------------------------------------
// Depthwise conv kxk stride s pad (k-1)/2, NHWC f16 in (stride Cpad) -> f32 out
// ---------------------------------------------------------------------------
__global__ void k_dwconv(const _Float16* __restrict__ x, const float* __restrict__ w,
                         float* __restrict__ y, int C, int Cpad, int H, int W,
                         int K, int S, int Ho, int Wo, int n) {
  int i = blockIdx.x * blockDim.x + threadIdx.x;
  if (i >= n) return;
  int c  = i % C;
  int px = i / C;
  int wo = px % Wo;
  int ho = px / Wo;
  int pad = (K - 1) / 2;
  const float* wp = w + (size_t)c * K * K;
  float acc = 0.f;
  for (int ky = 0; ky < K; ++ky) {
    int yy = ho * S - pad + ky;
    if ((unsigned)yy >= (unsigned)H) continue;
    for (int kx = 0; kx < K; ++kx) {
      int xx = wo * S - pad + kx;
      if ((unsigned)xx >= (unsigned)W) continue;
      acc += (float)x[((size_t)yy * W + xx) * Cpad + c] * wp[ky * K + kx];
    }
  }
  y[i] = acc;
}

// ---------------------------------------------------------------------------
// Per-channel mean/var over NHWC f32 (biased var, as jnp.var). Block/channel.
// ---------------------------------------------------------------------------
__global__ void k_bn_stats(const float* __restrict__ x, int C, int HW,
                           float* __restrict__ stats) {
  int c = blockIdx.x;
  float s = 0.f, ss = 0.f;
  for (int p = threadIdx.x; p < HW; p += blockDim.x) {
    float v = x[(size_t)p * C + c];
    s += v;
    ss += v * v;
  }
  __shared__ float sh[512];
  sh[threadIdx.x]       = s;
  sh[256 + threadIdx.x] = ss;
  __syncthreads();
  for (int o = 128; o > 0; o >>= 1) {
    if ((int)threadIdx.x < o) {
      sh[threadIdx.x]       += sh[threadIdx.x + o];
      sh[256 + threadIdx.x] += sh[256 + threadIdx.x + o];
    }
    __syncthreads();
  }
  if (threadIdx.x == 0) {
    float mean = sh[0] / (float)HW;
    stats[2 * c]     = mean;
    stats[2 * c + 1] = sh[256] / (float)HW - mean * mean;
  }
}

// normalize (+act) f32 NHWC stride C  ->  f16 NHWC stride Cpad (pads zeroed)
__global__ void k_bn_act_f16(const float* __restrict__ y, const float* __restrict__ stats,
                             const float* __restrict__ g, const float* __restrict__ b,
                             int C, int Cpad, int act, int n, _Float16* __restrict__ out) {
  int i = blockIdx.x * blockDim.x + threadIdx.x;
  if (i >= n) return;
  int c = i % Cpad;
  int p = i / Cpad;
  float v = 0.f;
  if (c < C) {
    v = (y[(size_t)p * C + c] - stats[2 * c]) * rsqrtf(stats[2 * c + 1] + 1e-3f) * g[c] + b[c];
    v = apply_act(v, act);
  }
  out[i] = (_Float16)v;
}

// per-channel spatial mean over NHWC f16 (stride Cpad). One block per channel.
__global__ void k_avgpool(const _Float16* __restrict__ x, int Cpad, int HW,
                          float* __restrict__ out) {
  int c = blockIdx.x;
  float s = 0.f;
  for (int p = threadIdx.x; p < HW; p += blockDim.x) s += (float)x[(size_t)p * Cpad + c];
  __shared__ float sh[256];
  sh[threadIdx.x] = s;
  __syncthreads();
  for (int o = 128; o > 0; o >>= 1) {
    if ((int)threadIdx.x < o) sh[threadIdx.x] += sh[threadIdx.x + o];
    __syncthreads();
  }
  if (threadIdx.x == 0) out[c] = sh[0] / (float)HW;
}

// y[o] = act(sum_i W[o][i]*x[i] + b[o])   (fp32, small)
__global__ void k_matvec(const float* __restrict__ W, const float* __restrict__ b,
                         const float* __restrict__ x, float* __restrict__ y,
                         int In, int Out, int act) {
  int o = blockIdx.x * blockDim.x + threadIdx.x;
  if (o >= Out) return;
  float acc = b ? b[o] : 0.f;
  const float* wr = W + (size_t)o * In;
  for (int i = 0; i < In; ++i) acc += wr[i] * x[i];
  y[o] = apply_act(acc, act);
}

__global__ void k_se_scale(_Float16* __restrict__ x, const float* __restrict__ w,
                           int C, int Cpad, int n) {
  int i = blockIdx.x * blockDim.x + threadIdx.x;
  if (i >= n) return;
  int c = i % Cpad;
  if (c < C) x[i] = (_Float16)((float)x[i] * w[c]);
}

__global__ void k_add_f16(_Float16* __restrict__ y, const _Float16* __restrict__ r, int n) {
  int i = blockIdx.x * blockDim.x + threadIdx.x;
  if (i >= n) return;
  y[i] = (_Float16)((float)y[i] + (float)r[i]);
}

// ---------------------------------------------------------------------------
// QP head: PDIPM, n=m=8, 20 iterations. Single thread (serial tiny work).
// ---------------------------------------------------------------------------
__global__ void k_qp(const float* __restrict__ f, const float* __restrict__ cam,
                     const float* __restrict__ vel, const float* __restrict__ ctrl,
                     const float* __restrict__ rot_w, const float* __restrict__ rot_b,
                     const float* __restrict__ A_w, const float* __restrict__ A_b,
                     const float* __restrict__ b_w, const float* __restrict__ b_b,
                     float* __restrict__ out) {
  if (threadIdx.x != 0 || blockIdx.x != 0) return;
  float cv[6];
  for (int i = 0; i < 3; ++i) { cv[i] = cam[i]; cv[3 + i] = vel[i]; }
  float R[8][8];
  for (int j = 0; j < 64; ++j) {
    float a = rot_b[j];
    for (int k = 0; k < 6; ++k) a += rot_w[j * 6 + k] * cv[k];
    R[j / 8][j % 8] = a;
  }
  float rot[8];
  for (int i = 0; i < 8; ++i) {
    float a = 0.f;
    for (int j = 0; j < 8; ++j) a += R[i][j] * f[j];
    rot[i] = fmaxf(a, 0.f);
  }
  float Aq[8][4];
  for (int j = 0; j < 32; ++j) {
    float a = A_b[j];
    for (int k = 0; k < 8; ++k) a += A_w[j * 8 + k] * rot[k];
    Aq[j / 4][j % 4] = fabsf(a);
  }
  float h[8];
  for (int j = 0; j < 8; ++j) {
    float a = b_b[j];
    for (int k = 0; k < 8; ++k) a += b_w[j * 8 + k] * rot[k];
    h[j] = a;
  }
  float G[8][8];
  for (int r = 0; r < 8; ++r)
    for (int c = 0; c < 4; ++c) { G[r][c] = Aq[r][c]; G[r][c + 4] = Aq[r][c]; }
  float q[8];
  for (int i = 0; i < 4; ++i) { q[i] = -ctrl[i]; q[4 + i] = ctrl[i]; }

  float z[8] = {0}, s[8], lam[8];
  for (int i = 0; i < 8; ++i) { s[i] = 1.f; lam[i] = 1.f; }
  const float BIG = 1e9f;
  for (int it = 0; it < 20; ++it) {
    float rd[8], rp[8];
    for (int i = 0; i < 8; ++i) {
      float a = z[i] + q[i];
      for (int m = 0; m < 8; ++m) a += G[m][i] * lam[m];
      rd[i] = a;
    }
    for (int m = 0; m < 8; ++m) {
      float a = s[m] - h[m];
      for (int i = 0; i < 8; ++i) a += G[m][i] * z[i];
      rp[m] = a;
    }
    float mu = 0.f;
    for (int m = 0; m < 8; ++m) mu += s[m] * lam[m];
    mu *= (1.f / 8.f);
    float t = 0.1f * mu;
    float d[8], w2[8];
    for (int m = 0; m < 8; ++m) {
      d[m]  = lam[m] / s[m];
      w2[m] = d[m] * rp[m] + (t - s[m] * lam[m]) / s[m];
    }
    float rhs[8];
    for (int i = 0; i < 8; ++i) {
      float a = -rd[i];
      for (int m = 0; m < 8; ++m) a -= G[m][i] * w2[m];
      rhs[i] = a;
    }
    float M[8][9];
    for (int i = 0; i < 8; ++i) {
      for (int j = 0; j < 8; ++j) {
        float a = (i == j) ? 1.f : 0.f;
        for (int m = 0; m < 8; ++m) a += G[m][i] * d[m] * G[m][j];
        M[i][j] = a;
      }
      M[i][8] = rhs[i];
    }
    for (int c = 0; c < 8; ++c) {
      int piv = c;
      float best = fabsf(M[c][c]);
      for (int r = c + 1; r < 8; ++r) {
        float v = fabsf(M[r][c]);
        if (v > best) { best = v; piv = r; }
      }
      if (piv != c)
        for (int k = c; k < 9; ++k) { float tm = M[c][k]; M[c][k] = M[piv][k]; M[piv][k] = tm; }
      float inv = 1.f / M[c][c];
      for (int r = 0; r < 8; ++r) {
        if (r == c) continue;
        float fac = M[r][c] * inv;
        for (int k = c; k < 9; ++k) M[r][k] -= fac * M[c][k];
      }
    }
    float dz[8];
    for (int i = 0; i < 8; ++i) dz[i] = M[i][8] / M[i][i];
    float Gdz[8];
    for (int m = 0; m < 8; ++m) {
      float a = 0.f;
      for (int i = 0; i < 8; ++i) a += G[m][i] * dz[i];
      Gdz[m] = a;
    }
    float dlam[8], ds[8];
    for (int m = 0; m < 8; ++m) {
      dlam[m] = d[m] * (Gdz[m] + rp[m]) + (t - s[m] * lam[m]) / s[m];
      ds[m]   = -rp[m] - Gdz[m];
    }
    float a1 = BIG, a2 = BIG;
    for (int m = 0; m < 8; ++m) {
      if (ds[m] < 0.f)   a1 = fminf(a1, -s[m] / ds[m]);
      if (dlam[m] < 0.f) a2 = fminf(a2, -lam[m] / dlam[m]);
    }
    float al = fminf(1.f, 0.99f * fminf(a1, a2));
    for (int i = 0; i < 8; ++i) z[i] += al * dz[i];
    for (int m = 0; m < 8; ++m) { s[m] += al * ds[m]; lam[m] += al * dlam[m]; }
  }
  float y0 = 0.5f * (z[0] + z[4]), y1 = 0.5f * (z[1] + z[5]);
  float y2 = 0.5f * (z[2] + z[6]), y3 = 0.5f * (z[3] + z[7]);
  out[0] = fminf(fmaxf(y0, -1.f), 1.f);
  out[1] = fminf(fmaxf(y1, -1.f), 1.f);
  out[2] = fminf(fmaxf(y2, 0.f), 1.f);
  out[3] = fminf(fmaxf(y3, 0.f), 1.f);
}

// ============================================================================
// Host orchestration.  Input index map = JAX pytree flattening (dict keys
// sorted): 0=camera_direction, 1=controller_input, 2=image, 3..154 params
// leaves, 155=relative_velocity. Blocks at base 9,19,28,37,50,63,76,89,102,
// 115,128; cls1_b=141 cls1_w=142 cls2_b=143 cls2_w=144 last_b/g/w=145..147
// rescale_b/w=148,149 rot_b/w=150,151 stem_b/g/w=152..154.
// ============================================================================
static inline int pad32(int c) { return (c + 31) & ~31; }

extern "C" void kernel_launch(void* const* d_in, const int* in_sizes, int n_in,
                              void* d_out, int out_size, void* d_ws, size_t ws_size,
                              hipStream_t stream) {
  (void)in_sizes; (void)n_in; (void)out_size; (void)ws_size;
  auto in = [&](int i) { return (const float*)d_in[i]; };

  // ---- workspace carve-up ---------------------------------------------------
  const size_t R32N = 3600000;   // rescale NHWC f32 (3*768*1536 = 3,538,944)
  const size_t Y32N = 5600000;   // raw conv output f32 (max 73728*72)
  const size_t H16N = 9600000;   // f16 activation buffers (max 294912*32)
  float*    R32  = (float*)d_ws;
  float*    Yb   = R32 + R32N;
  _Float16* Xh   = (_Float16*)(Yb + Y32N);
  _Float16* Th   = Xh + H16N;
  _Float16* Dh   = Th + H16N;
  _Float16* Whb  = Dh + H16N;          // padded f16 weights (max 576*96)
  float*    stats  = (float*)(Whb + 65536);
  float*    pooled = stats + 1200;
  float*    sev    = pooled + 640;
  float*    sew    = sev + 160;
  float*    fc1    = sew + 640;
  float*    fc2    = fc1 + 1024;
  float*    fvec   = fc2 + 1024;

  // conv(1x1) -> BN(train) -> act -> f16 NHWC (stride pad32(Cout))
  auto convbn = [&](const _Float16* Xi, int CinPad, _Float16* Oh, const float* Wt,
                    const float* g, const float* bb, int Cin, int Cout, int HW, int act) {
    int Cout16 = (Cout + 15) & ~15;
    int wn = Cout16 * CinPad;
    k_w2h<<<(wn + 255) / 256, 256, 0, stream>>>(Wt, Whb, Cout, Cin, CinPad, wn);
    dim3 grid((HW + 255) / 256, Cout16 / 16);
    k_gemm_wmma<<<grid, 256, 0, stream>>>(Xi, Whb, Yb, CinPad, Cout, HW);
    k_bn_stats<<<Cout, 256, 0, stream>>>(Yb, Cout, HW, stats);
    int Cpo = pad32(Cout);
    int n = HW * Cpo;
    k_bn_act_f16<<<(n + 255) / 256, 256, 0, stream>>>(Yb, stats, g, bb, Cout, Cpo, act,
                                                      n, Oh);
  };

  // 1) rescale -> R32 (NHWC f32)
  {
    int n = 3 * 768 * 1536;
    k_rescale<<<(n + 255) / 256, 256, 0, stream>>>(in(2), in(149), in(148), R32, n);
  }
  // 2) stem -> Yb (NHWC f32, C=16); BN+hswish -> Xh (Cpad=32)
  {
    int HW = 384 * 768, n = HW * 16;
    k_stem<<<(n + 255) / 256, 256, 0, stream>>>(R32, in(154), Yb, n);
    k_bn_stats<<<16, 256, 0, stream>>>(Yb, 16, HW, stats);
    int np = HW * 32;
    k_bn_act_f16<<<(np + 255) / 256, 256, 0, stream>>>(Yb, stats, in(153), in(152), 16,
                                                       32, 2, np, Xh);
  }

  _Float16* X = Xh;  // current block input (f16, stride pad32(cin))
  _Float16* T = Th;  // expand / project output
  _Float16* D = Dh;  // depthwise output
  int H = 384, W = 768;

  struct Blk { int cin, k, exp, out, se, act, s, base, sq; };
  const Blk blks[11] = {
      {16, 3,  16, 16, 1, 1, 2,   9,   8},
      {16, 3,  72, 24, 0, 1, 2,  19,   0},
      {24, 3,  88, 24, 0, 1, 1,  28,   0},
      {24, 5,  96, 40, 1, 2, 2,  37,  24},
      {40, 5, 240, 40, 1, 2, 1,  50,  64},
      {40, 5, 240, 40, 1, 2, 1,  63,  64},
      {40, 5, 120, 48, 1, 2, 1,  76,  32},
      {48, 5, 144, 48, 1, 2, 1,  89,  40},
      {48, 5, 288, 96, 1, 2, 2, 102,  72},
      {96, 5, 576, 96, 1, 2, 1, 115, 144},
      {96, 5, 576, 96, 1, 2, 1, 128, 144},
  };

  for (int i = 0; i < 11; ++i) {
    const Blk& B = blks[i];
    bool has_pw = (B.exp != B.cin);
    int idw = B.base;
    int ipj = B.base + 3;
    int ipw = has_pw ? B.base + 6 : -1;
    int ise = B.se ? (has_pw ? B.base + 9 : B.base + 6) : -1;
    int HWi = H * W;
    int cinP = pad32(B.cin), expP = pad32(B.exp), outP = pad32(B.out);

    const _Float16* x = X;
    int xP = cinP;
    if (has_pw) {
      convbn(X, cinP, T, in(ipw + 2), in(ipw + 1), in(ipw + 0), B.cin, B.exp, HWi, B.act);
      x = T;
      xP = expP;
    }
    int pad = (B.k - 1) / 2;
    int Ho = (H + 2 * pad - B.k) / B.s + 1;
    int Wo = (W + 2 * pad - B.k) / B.s + 1;
    int HWo = Ho * Wo;
    int nd = HWo * B.exp;
    k_dwconv<<<(nd + 255) / 256, 256, 0, stream>>>(x, in(idw + 2), Yb, B.exp, xP, H, W,
                                                   B.k, B.s, Ho, Wo, nd);
    k_bn_stats<<<B.exp, 256, 0, stream>>>(Yb, B.exp, HWo, stats);
    int ndp = HWo * expP;
    k_bn_act_f16<<<(ndp + 255) / 256, 256, 0, stream>>>(Yb, stats, in(idw + 1),
                                                        in(idw + 0), B.exp, expP, B.act,
                                                        ndp, D);
    if (B.se) {
      k_avgpool<<<B.exp, 256, 0, stream>>>(D, expP, HWo, pooled);
      k_matvec<<<(B.sq + 63) / 64, 64, 0, stream>>>(in(ise + 1), in(ise + 0), pooled,
                                                    sev, B.exp, B.sq, 1);
      k_matvec<<<(B.exp + 63) / 64, 64, 0, stream>>>(in(ise + 3), in(ise + 2), sev, sew,
                                                     B.sq, B.exp, 3);
      k_se_scale<<<(ndp + 255) / 256, 256, 0, stream>>>(D, sew, B.exp, expP, ndp);
    }
    // project (no activation) -> T (old pw buffer is dead by now)
    convbn(D, expP, T, in(ipj + 2), in(ipj + 1), in(ipj + 0), B.exp, B.out, HWo, 0);
    if (B.s == 1 && B.cin == B.out) {
      int n = HWo * outP;
      k_add_f16<<<(n + 255) / 256, 256, 0, stream>>>(T, X, n);
    }
    _Float16* tmp = X; X = T; T = tmp;
    H = Ho; W = Wo;
  }

  // last conv 96->576 + BN + hswish -> T (Cpad=576), then GAP
  convbn(X, pad32(96), T, in(147), in(146), in(145), 96, 576, H * W, 2);
  k_avgpool<<<576, 256, 0, stream>>>(T, 576, H * W, pooled);

  // classifier 576 -> 1024 (hswish) -> 1000 -> adapter 8
  k_matvec<<<(1024 + 63) / 64, 64, 0, stream>>>(in(142), in(141), pooled, fc1, 576, 1024, 2);
  k_matvec<<<(1000 + 63) / 64, 64, 0, stream>>>(in(144), in(143), fc1, fc2, 1024, 1000, 0);
  k_matvec<<<1, 64, 0, stream>>>(in(6), in(5), fc2, fvec, 1000, 8, 0);

  // QP head -> 4 outputs
  k_qp<<<1, 1, 0, stream>>>(fvec, in(0), in(155), in(1), in(151), in(150), in(4), in(3),
                            in(8), in(7), (float*)d_out);
}
